// CVLFuser_57217554317660
// MI455X (gfx1250) — matrix operand order repaired and checked by reading, not compiled, not run on hardware
//
#include <hip/hip_runtime.h>

// ---------------------------------------------------------------------------
// CDNA5 / gfx1250 WMMA types & op
// ---------------------------------------------------------------------------
typedef __attribute__((ext_vector_type(16))) __bf16        v16bf;
typedef __attribute__((ext_vector_type(8)))  float         v8f;
typedef __attribute__((ext_vector_type(8)))  unsigned int  v8u;

#define WMMA_BF16(a, b, c) \
    __builtin_amdgcn_wmma_f32_16x16x32_bf16(false, (a), false, (b), (short)0, (c), false, false)

constexpr int D_DIM   = 1024;
constexpr int M_ROWS  = 4096;
constexpr int NKB     = 65536;
constexpr int NCHUNK  = 16;
constexpr int CHUNK   = NKB / NCHUNK;   // 4096 kb rows per chunk
constexpr int BM      = 128;            // rows per workgroup
constexpr int BN      = 64;             // kb rows per inner tile
constexpr int BK      = 64;             // K tile
constexpr int LDA     = BK + 8;         // padded bf16 stride (bank-conflict free)
constexpr int LDB     = BK + 8;
constexpr int TOPK    = 32;
constexpr int THREADS = 256;            // 8 waves (wave32)

// ---------------------------------------------------------------------------
// bf16 split helpers:  x ~= hi + lo  (each bf16, round-to-nearest-even)
// ---------------------------------------------------------------------------
__device__ __forceinline__ unsigned short bf16_rn(float x) {
    unsigned int u = __float_as_uint(x);
    u += 0x7FFFu + ((u >> 16) & 1u);
    return (unsigned short)(u >> 16);
}
__device__ __forceinline__ float bf16_to_f(unsigned short h) {
    return __uint_as_float(((unsigned int)h) << 16);
}
__device__ __forceinline__ void split2(float x, unsigned short& h, unsigned short& l) {
    h = bf16_rn(x);
    l = bf16_rn(x - bf16_to_f(h));
}
// convert 4 f32 -> hi/lo bf16 and store 8B each into LDS
__device__ __forceinline__ void cvt_store4(const float4 v, unsigned short* hp, unsigned short* lp) {
    unsigned short h0, h1, h2, h3, l0, l1, l2, l3;
    split2(v.x, h0, l0); split2(v.y, h1, l1); split2(v.z, h2, l2); split2(v.w, h3, l3);
    uint2 H, L;
    H.x = (unsigned)h0 | ((unsigned)h1 << 16); H.y = (unsigned)h2 | ((unsigned)h3 << 16);
    L.x = (unsigned)l0 | ((unsigned)l1 << 16); L.y = (unsigned)l2 | ((unsigned)l3 << 16);
    *(uint2*)hp = H; *(uint2*)lp = L;
}

// ---------------------------------------------------------------------------
// WMMA fragment loads from LDS (per ISA 7.12.2 layouts, wave32)
// A 16x32 bf16: lane(half h, row) holds K in {kb+8h..kb+8h+7} and {kb+16+8h..}
// B 32x16 bf16: lane(half h, col) holds K in {kb+8h..kb+8h+7} and {kb+16+8h..}
// ---------------------------------------------------------------------------
__device__ __forceinline__ v16bf fragA(const unsigned short* rowp, int kbase, int half) {
    const uint4 a = *(const uint4*)(rowp + kbase + half * 8);
    const uint4 b = *(const uint4*)(rowp + kbase + 16 + half * 8);
    v8u t = {a.x, a.y, a.z, a.w, b.x, b.y, b.z, b.w};
    return __builtin_bit_cast(v16bf, t);
}
__device__ __forceinline__ v16bf fragB(const unsigned short* rowp, int kbase, int half) {
    const uint4 a = *(const uint4*)(rowp + kbase + half * 8);
    const uint4 b = *(const uint4*)(rowp + kbase + 16 + half * 8);
    v8u t = {a.x, a.y, a.z, a.w, b.x, b.y, b.z, b.w};
    return __builtin_bit_cast(v16bf, t);
}

__device__ __forceinline__ v8f vzero8() {
    v8f r;
#pragma unroll
    for (int i = 0; i < 8; ++i) r[i] = 0.0f;
    return r;
}

// One 64-K LDS tile worth of split-bf16 WMMAs for this wave's 2x2 tiles.
__device__ __forceinline__ void mma_tile(
    const unsigned short (*Ahi)[LDA], const unsigned short (*Alo)[LDA],
    const unsigned short (*Bhi)[LDB], const unsigned short (*Blo)[LDB],
    int wr, int wc, int lrow, int half, v8f acc[2][2])
{
#pragma unroll
    for (int s = 0; s < 2; ++s) {
        const int kb = s * 32;
        v16bf aH[2], aL[2], bH[2], bL[2];
#pragma unroll
        for (int ti = 0; ti < 2; ++ti) {
            aH[ti] = fragA(Ahi[(wr * 2 + ti) * 16 + lrow], kb, half);
            aL[ti] = fragA(Alo[(wr * 2 + ti) * 16 + lrow], kb, half);
        }
#pragma unroll
        for (int tj = 0; tj < 2; ++tj) {
            bH[tj] = fragB(Bhi[(wc * 2 + tj) * 16 + lrow], kb, half);
            bL[tj] = fragB(Blo[(wc * 2 + tj) * 16 + lrow], kb, half);
        }
#pragma unroll
        for (int ti = 0; ti < 2; ++ti)
#pragma unroll
            for (int tj = 0; tj < 2; ++tj) {
                acc[ti][tj] = WMMA_BF16(aH[ti], bH[tj], acc[ti][tj]);   // hi*hi
                acc[ti][tj] = WMMA_BF16(aH[ti], bL[tj], acc[ti][tj]);   // hi*lo
                acc[ti][tj] = WMMA_BF16(aL[ti], bH[tj], acc[ti][tj]);   // lo*hi
            }
    }
}

// ---------------------------------------------------------------------------
// Kernel 1: kb row squared norms (one wave per row)
// ---------------------------------------------------------------------------
__global__ __launch_bounds__(THREADS) void kbsq_kernel(const float* __restrict__ kb,
                                                       float* __restrict__ kbsq) {
    const int wave = threadIdx.x >> 5, lane = threadIdx.x & 31;
    const int row  = blockIdx.x * 8 + wave;
    const float* p = kb + (size_t)row * D_DIM;
    float acc = 0.0f;
    for (int i = lane; i < D_DIM; i += 32) { const float v = p[i]; acc += v * v; }
#pragma unroll
    for (int o = 16; o > 0; o >>= 1) acc += __shfl_down(acc, o, 32);
    if (lane == 0) kbsq[row] = acc;
}

// ---------------------------------------------------------------------------
// Kernel 2: q = C @ Qw^T (split-bf16 WMMA); emit q as bf16 hi/lo planes + |q|^2
// ---------------------------------------------------------------------------
__global__ __launch_bounds__(THREADS) void qgemm_kernel(
    const float* __restrict__ Cm, const float* __restrict__ Qw,
    unsigned short* __restrict__ qhi, unsigned short* __restrict__ qlo,
    float* __restrict__ qsq)
{
    __shared__ unsigned short Ahi[BM][LDA], Alo[BM][LDA];
    __shared__ unsigned short Bhi[BN][LDB], Blo[BN][LDB];
    __shared__ float rowSq[BM];

    const int tid = threadIdx.x;
    const int lane = tid & 31, wave = tid >> 5;
    const int half = lane >> 4, lrow = lane & 15;
    const int wr = wave >> 1, wc = wave & 1;
    const int rowBase = blockIdx.x * BM;
    const int nBase   = blockIdx.y * BN;

    if (tid < BM) rowSq[tid] = 0.0f;

    v8f acc[2][2];
    acc[0][0] = vzero8(); acc[0][1] = vzero8(); acc[1][0] = vzero8(); acc[1][1] = vzero8();

#pragma unroll 1
    for (int kt = 0; kt < D_DIM / BK; ++kt) {
        const int k0 = kt * BK;
        // A tile: 128x64 f32 from C -> split bf16
#pragma unroll
        for (int i = 0; i < 8; ++i) {
            const int c = tid + THREADS * i;        // 2048 float4 chunks
            const int r = c >> 4, cc = (c & 15) << 2;
            const float4 v = *(const float4*)&Cm[(size_t)(rowBase + r) * D_DIM + k0 + cc];
            cvt_store4(v, &Ahi[r][cc], &Alo[r][cc]);
        }
        // B tile: 64x64 f32 from Qw -> split bf16
#pragma unroll
        for (int i = 0; i < 4; ++i) {
            const int c = tid + THREADS * i;        // 1024 float4 chunks
            const int r = c >> 4, cc = (c & 15) << 2;
            const float4 v = *(const float4*)&Qw[(size_t)(nBase + r) * D_DIM + k0 + cc];
            cvt_store4(v, &Bhi[r][cc], &Blo[r][cc]);
        }
        __syncthreads();
        mma_tile(Ahi, Alo, Bhi, Blo, wr, wc, lrow, half, acc);
        __syncthreads();
    }

    // epilogue: store q hi/lo, accumulate |q|^2
#pragma unroll
    for (int ti = 0; ti < 2; ++ti)
#pragma unroll
        for (int tj = 0; tj < 2; ++tj)
#pragma unroll
            for (int r = 0; r < 8; ++r) {
                const int m = (wr * 2 + ti) * 16 + r + half * 8;
                const int n = (wc * 2 + tj) * 16 + lrow;
                const float q = acc[ti][tj][r];
                unsigned short h, l; split2(q, h, l);
                const size_t g = (size_t)(rowBase + m) * D_DIM + nBase + n;
                qhi[g] = h; qlo[g] = l;
                atomicAdd(&rowSq[m], q * q);
            }
    __syncthreads();
    if (tid < BM) atomicAdd(&qsq[rowBase + tid], rowSq[tid]);
}

// ---------------------------------------------------------------------------
// Kernel 3: fused distance GEMM (split-bf16 WMMA) + streaming per-row top-32.
// Software-pipelined: next K-tile is fetched into ping-pong register buffers
// while the current tile's WMMAs run; kb stream prefetched 2 tiles ahead.
// grid = (M/BM, NCHUNK); each WG scans CHUNK kb rows for its 128 q rows.
// ---------------------------------------------------------------------------
__global__ __launch_bounds__(THREADS) void dist_topk_kernel(
    const float* __restrict__ kb,
    const unsigned short* __restrict__ qhi, const unsigned short* __restrict__ qlo,
    const float* __restrict__ qsq, const float* __restrict__ kbsq,
    float* __restrict__ pD, unsigned int* __restrict__ pI)
{
    __shared__ unsigned short Ahi[BM][LDA], Alo[BM][LDA];
    __shared__ unsigned short Bhi[BN][LDB], Blo[BN][LDB];
    __shared__ float         stage[BM][BN + 1];
    __shared__ float         topD[BM][TOPK + 1];
    __shared__ unsigned int  topI[BM][TOPK + 1];
    __shared__ float qsqL[BM];
    __shared__ float kbsqL[BN];

    const int tid = threadIdx.x;
    const int lane = tid & 31, wave = tid >> 5;
    const int half = lane >> 4, lrow = lane & 15;
    const int wr = wave >> 1, wc = wave & 1;
    const int rowBase = blockIdx.x * BM;
    const int colBase = blockIdx.y * CHUNK;

    if (tid < BM) qsqL[tid] = qsq[rowBase + tid];

    // per-thread load/store helpers for one K-tile
    auto loadA = [&](int it, uint4* aH, uint4* aL) {
        const int k0 = (it & 15) * BK;
#pragma unroll
        for (int i = 0; i < 4; ++i) {
            const int c = tid + THREADS * i;        // 1024 chunks of 8 ushort
            const int r = c >> 3, cc = (c & 7) << 3;
            const size_t g = (size_t)(rowBase + r) * D_DIM + k0 + cc;
            aH[i] = *(const uint4*)&qhi[g];
            aL[i] = *(const uint4*)&qlo[g];
        }
    };
    auto loadB = [&](int it, float4* b) {
        const int nB = colBase + (it >> 4) * BN;
        const int k0 = (it & 15) * BK;
#pragma unroll
        for (int i = 0; i < 4; ++i) {
            const int c = tid + THREADS * i;        // 1024 float4 chunks
            const int r = c >> 4, cc = (c & 15) << 2;
            b[i] = *(const float4*)&kb[(size_t)(nB + r) * D_DIM + k0 + cc];
        }
    };
    auto storeTiles = [&](const uint4* aH, const uint4* aL, const float4* b) {
#pragma unroll
        for (int i = 0; i < 4; ++i) {
            const int c = tid + THREADS * i;
            const int r = c >> 3, cc = (c & 7) << 3;
            *(uint4*)&Ahi[r][cc] = aH[i];
            *(uint4*)&Alo[r][cc] = aL[i];
        }
#pragma unroll
        for (int i = 0; i < 4; ++i) {
            const int c = tid + THREADS * i;
            const int r = c >> 4, cc = (c & 15) << 2;
            cvt_store4(b[i], &Bhi[r][cc], &Blo[r][cc]);
        }
    };

    // per-owner (thread t < BM owns row t) running top-32 state
    float maxv = 3.0e38f;
    int   maxp = 0, cnt = 0;

    constexpr int NIT = (CHUNK / BN) * (D_DIM / BK);   // 1024 K-tiles per chunk

    uint4  paH[2][4], paL[2][4];
    float4 pb[2][4];
    loadA(0, paH[0], paL[0]);
    loadB(0, pb[0]);

    v8f acc[2][2];

#pragma unroll 2                                       // makes cb/nb compile-time
    for (int it = 0; it < NIT; ++it) {
        const int cb = it & 1, nb = cb ^ 1;
        const int kt = it & 15;
        const int nBase = colBase + (it >> 4) * BN;

        __syncthreads();                               // LDS tile fully consumed
        if (kt == 0 && tid < BN) kbsqL[tid] = kbsq[nBase + tid];
        storeTiles(paH[cb], paL[cb], pb[cb]);
        if (it + 1 < NIT) {                            // prefetch next K-tile
            loadA(it + 1, paH[nb], paL[nb]);
            loadB(it + 1, pb[nb]);
        }
        if (it + 2 < NIT) {                            // gfx1250 global_prefetch on kb stream
            const int it2 = it + 2;
            const size_t g = (size_t)(colBase + (it2 >> 4) * BN + (tid >> 4)) * D_DIM +
                             (it2 & 15) * BK + ((tid & 15) << 2);
            __builtin_prefetch(&kb[g], 0, 1);
        }
        __syncthreads();                               // stores visible

        if (kt == 0) {
            acc[0][0] = vzero8(); acc[0][1] = vzero8();
            acc[1][0] = vzero8(); acc[1][1] = vzero8();
        }
        mma_tile(Ahi, Alo, Bhi, Blo, wr, wc, lrow, half, acc);

        if (kt == 15) {
            // d2 = |q|^2 + |kb|^2 - 2 q.kb  (clamped) -> stage
#pragma unroll
            for (int ti = 0; ti < 2; ++ti)
#pragma unroll
                for (int tj = 0; tj < 2; ++tj)
#pragma unroll
                    for (int r = 0; r < 8; ++r) {
                        const int m = (wr * 2 + ti) * 16 + r + half * 8;
                        const int n = (wc * 2 + tj) * 16 + lrow;
                        float d2 = qsqL[m] + kbsqL[n] - 2.0f * acc[ti][tj][r];
                        stage[m][n] = fmaxf(d2, 0.0f);
                    }
            __syncthreads();

            // streaming top-32 update (owner thread per row)
            if (tid < BM) {
                const int t = tid;
#pragma unroll 1
                for (int j = 0; j < BN; ++j) {
                    const float d = stage[t][j];
                    const unsigned int idx = (unsigned int)(nBase + j);
                    if (cnt < TOPK) {
                        topD[t][cnt] = d; topI[t][cnt] = idx;
                        if (cnt == 0 || d > maxv) { maxv = d; maxp = cnt; }
                        ++cnt;
                    } else if (d < maxv) {             // ties keep earlier (lower) index
                        topD[t][maxp] = d; topI[t][maxp] = idx;
                        float mv = -1.0f; int mp = 0;
#pragma unroll
                        for (int i = 0; i < TOPK; ++i) {
                            const float v = topD[t][i];
                            if (v > mv) { mv = v; mp = i; }
                        }
                        maxv = mv; maxp = mp;
                    }
                }
            }
            // no trailing barrier needed: loop-top barrier orders next stores
        }
    }

    if (tid < BM) {   // top lists are owner-private; safe without a barrier
        const size_t base = ((size_t)(rowBase + tid) * NCHUNK + blockIdx.y) * TOPK;
#pragma unroll 1
        for (int i = 0; i < TOPK; ++i) { pD[base + i] = topD[tid][i]; pI[base + i] = topI[tid][i]; }
    }
}

// ---------------------------------------------------------------------------
// Kernel 4: per row merge 16x32 partials -> exact top-32 -> softmax -> gather
// and write all three silu'd output segments.
// ---------------------------------------------------------------------------
__device__ __forceinline__ float silu(float x) { return x / (1.0f + expf(-x)); }

__global__ __launch_bounds__(256) void merge_retrieve_kernel(
    const float* __restrict__ Cm, const float* __restrict__ Km,
    const float* __restrict__ kb, const float* __restrict__ tempPtr,
    const float* __restrict__ pD, const unsigned int* __restrict__ pI,
    float* __restrict__ out)
{
    constexpr int NCAND = NCHUNK * TOPK;  // 512
    __shared__ float        dL[NCAND];
    __shared__ unsigned int iL[NCAND];
    __shared__ float        selD[TOPK];
    __shared__ unsigned int selI[TOPK];
    __shared__ float        wL[TOPK];
    __shared__ float        wsum;

    const int tid = threadIdx.x;
    const int row = blockIdx.x;
    const size_t cbase = (size_t)row * NCAND;

    dL[tid]       = pD[cbase + tid];       iL[tid]       = pI[cbase + tid];
    dL[tid + 256] = pD[cbase + tid + 256]; iL[tid + 256] = pI[cbase + tid + 256];
    __syncthreads();

    // exact rank selection (tie-break: lower kb index, matching jax top_k)
#pragma unroll 1
    for (int c = tid; c < NCAND; c += 256) {
        const float d = dL[c]; const unsigned int id = iL[c];
        int rank = 0;
#pragma unroll 4
        for (int j = 0; j < NCAND; ++j) {
            const float dj = dL[j]; const unsigned int ij = iL[j];
            rank += (dj < d) || (dj == d && ij < id);
        }
        if (rank < TOPK) { selD[rank] = sqrtf(d); selI[rank] = id; }
    }
    __syncthreads();

    const float invT = 1.0f / tempPtr[0];
    if (tid < TOPK) wL[tid] = expf(-(selD[tid] - selD[0]) * invT);  // selD[0] = min dist
    __syncthreads();
    if (tid == 0) {
        float s = 0.0f;
#pragma unroll
        for (int j = 0; j < TOPK; ++j) s += wL[j];
        wsum = s;
    }
    __syncthreads();
    const float inv = 1.0f / wsum;

    const int c0 = tid * 4;
    float4 a = {0.0f, 0.0f, 0.0f, 0.0f};
#pragma unroll 1
    for (int j = 0; j < TOPK; ++j) {
        const float w = wL[j];
        const float4 v = *(const float4*)&kb[(size_t)selI[j] * D_DIM + c0];
        a.x += w * v.x; a.y += w * v.y; a.z += w * v.z; a.w += w * v.w;
    }
    const size_t ob = (size_t)row * 3072;
    float4 o;
    o.x = silu(0.25f * a.x * inv); o.y = silu(0.25f * a.y * inv);
    o.z = silu(0.25f * a.z * inv); o.w = silu(0.25f * a.w * inv);
    *(float4*)&out[ob + 2048 + c0] = o;

    const float4 cv = *(const float4*)&Cm[(size_t)row * D_DIM + c0];
    o.x = silu(cv.x); o.y = silu(cv.y); o.z = silu(cv.z); o.w = silu(cv.w);
    *(float4*)&out[ob + c0] = o;

    const float4 kv = *(const float4*)&Km[(size_t)row * D_DIM + c0];
    o.x = silu(0.5f * kv.x); o.y = silu(0.5f * kv.y);
    o.z = silu(0.5f * kv.z); o.w = silu(0.5f * kv.w);
    *(float4*)&out[ob + 1024 + c0] = o;
}

// ---------------------------------------------------------------------------
// Host-side launcher
// ---------------------------------------------------------------------------
extern "C" void kernel_launch(void* const* d_in, const int* in_sizes, int n_in,
                              void* d_out, int out_size, void* d_ws, size_t ws_size,
                              hipStream_t stream) {
    (void)in_sizes; (void)n_in; (void)out_size; (void)ws_size;
    const float* Cm   = (const float*)d_in[0];
    const float* Km   = (const float*)d_in[1];
    const float* kb   = (const float*)d_in[2];
    const float* Qw   = (const float*)d_in[3];
    const float* temp = (const float*)d_in[4];   // top_k (d_in[5]) assumed == 32

    char* ws = (char*)d_ws;
    size_t off = 0;
    auto carve = [&](size_t bytes) {
        void* p = ws + off;
        off = (off + bytes + 255) & ~(size_t)255;
        return p;
    };
    unsigned short* qhi  = (unsigned short*)carve((size_t)M_ROWS * D_DIM * 2);
    unsigned short* qlo  = (unsigned short*)carve((size_t)M_ROWS * D_DIM * 2);
    float*          qsq  = (float*)carve((size_t)M_ROWS * 4);
    float*          kbsq = (float*)carve((size_t)NKB * 4);
    float*          pD   = (float*)carve((size_t)M_ROWS * NCHUNK * TOPK * 4);
    unsigned int*   pI   = (unsigned int*)carve((size_t)M_ROWS * NCHUNK * TOPK * 4);

    hipMemsetAsync(qsq, 0, (size_t)M_ROWS * 4, stream);

    kbsq_kernel<<<NKB / 8, THREADS, 0, stream>>>(kb, kbsq);
    qgemm_kernel<<<dim3(M_ROWS / BM, D_DIM / BN), THREADS, 0, stream>>>(Cm, Qw, qhi, qlo, qsq);
    dist_topk_kernel<<<dim3(M_ROWS / BM, NCHUNK), THREADS, 0, stream>>>(kb, qhi, qlo, qsq, kbsq, pD, pI);
    merge_retrieve_kernel<<<M_ROWS, 256, 0, stream>>>(Cm, Km, kb, temp, pD, pI, (float*)d_out);
}
